// BERT_6133213299297
// MI455X (gfx1250) — matrix operand order repaired and checked by reading, not compiled, not run on hardware
//
#include <hip/hip_runtime.h>

typedef _Float16 h8  __attribute__((ext_vector_type(8)));
typedef _Float16 h16 __attribute__((ext_vector_type(16)));
typedef float    f8  __attribute__((ext_vector_type(8)));

#define WMMA_F32_F16(a, b, c) \
  __builtin_amdgcn_wmma_f32_16x16x32_f16(false, (a), false, (b), (short)0, (c), false, false)

static __device__ __forceinline__ h16 join8(h8 a, h8 b) {
  return __builtin_shufflevector(a, b, 0, 1, 2, 3, 4, 5, 6, 7, 8, 9, 10, 11, 12, 13, 14, 15);
}

// Low 32 bits of a generic pointer to __shared__ = LDS byte address
// (LDS aperture only occupies the high word; hardware truncates to addr[31:0]).
static __device__ __forceinline__ unsigned lds_addr(const void* p) {
  return (unsigned)(unsigned long long)p;
}

// 16B-per-lane async copy global -> LDS, tracked by ASYNCcnt.
static __device__ __forceinline__ void async_copy16(const void* g, unsigned lds) {
  unsigned long long ga = (unsigned long long)g;
  asm volatile("global_load_async_to_lds_b128 %0, %1, off"
               :: "v"(lds), "v"(ga) : "memory");
}

// ---------------------------------------------------------------------------
// Kernel 1: fused QKV projection (x @ W^T + b), RoPE on q/k, scale q by sqrt(64).
// f16 outputs: q,k as [B,H,S,64]; v pre-transposed as [B,H,64,S].
// M = B*S = 4096, N = 1024, K = 1024. Block tile 128x128, BK = 32.
// 8 waves: 4 (M) x 2 (N); wave tile 32x64 -> 2x4 WMMA accumulators.
// ---------------------------------------------------------------------------
__global__ __launch_bounds__(256) void qkv_rope_kernel(
    const float* __restrict__ X,
    const float* __restrict__ Wq, const float* __restrict__ bq,
    const float* __restrict__ Wk, const float* __restrict__ bk,
    const float* __restrict__ Wv, const float* __restrict__ bv,
    _Float16* __restrict__ qws, _Float16* __restrict__ kws,
    _Float16* __restrict__ vws) {
  constexpr int K = 1024, S = 2048;
  const int z = blockIdx.z;
  const float* Wm = (z == 0) ? Wq : (z == 1) ? Wk : Wv;
  const float* bm = (z == 0) ? bq : (z == 1) ? bk : bv;

  const int m0 = blockIdx.x * 128;
  const int n0 = blockIdx.y * 128;
  const int tid = threadIdx.x;
  const int lane = tid & 31;
  const int ln = lane & 15;
  const int lh = lane >> 4;
  const int wave = tid >> 5;  // 0..7 (wave32)
  const int wm = wave >> 1;   // 0..3 -> 32 rows each
  const int wn = wave & 1;    // 0..1 -> 64 cols each

  __shared__ __align__(32) _Float16 As[128 * 32];
  __shared__ __align__(32) _Float16 Bs[128 * 32];

  f8 acc[2][4] = {};

  for (int kt = 0; kt < K; kt += 32) {
    __syncthreads();
    // Stage 128x32 f32 -> f16 tiles of X and W (both K-major rows).
    for (int i = 0; i < 4; ++i) {
      int vec = tid + 256 * i;  // 0..1023 float4s
      int row = vec >> 3;       // 0..127
      int kc = (vec & 7) * 4;
      float4 av = *(const float4*)&X[(size_t)(m0 + row) * K + kt + kc];
      float4 wv = *(const float4*)&Wm[(size_t)(n0 + row) * K + kt + kc];
      _Float16* ap = &As[row * 32 + kc];
      ap[0] = (_Float16)av.x; ap[1] = (_Float16)av.y;
      ap[2] = (_Float16)av.z; ap[3] = (_Float16)av.w;
      _Float16* bp = &Bs[row * 32 + kc];
      bp[0] = (_Float16)wv.x; bp[1] = (_Float16)wv.y;
      bp[2] = (_Float16)wv.z; bp[3] = (_Float16)wv.w;
    }
    __syncthreads();

    // A fragments: lane holds row m, K chunks [kb8..kb8+7] and [kb8+16..kb8+23].
    h16 afr[2];
    for (int i = 0; i < 2; ++i) {
      int m = wm * 32 + i * 16 + ln;
      int kb8 = lh * 8;
      const h8* p0 = (const h8*)&As[m * 32 + kb8];
      const h8* p1 = (const h8*)&As[m * 32 + kb8 + 16];
      afr[i] = join8(*p0, *p1);
    }
    // B fragments: lane holds col n, K chunks [lh*8..lh*8+7] and [lh*8+16..lh*8+23].  B[k][n] = W[n][k].
    h16 bfr[4];
    for (int j = 0; j < 4; ++j) {
      int n = wn * 64 + j * 16 + ln;
      bfr[j] = join8(*(const h8*)&Bs[n * 32 + lh * 8], *(const h8*)&Bs[n * 32 + lh * 8 + 16]);
    }
    for (int i = 0; i < 2; ++i)
      for (int j = 0; j < 4; ++j)
        acc[i][j] = WMMA_F32_F16(afr[i], bfr[j], acc[i][j]);
  }

  // Bias (before RoPE, matching reference order).
  for (int j = 0; j < 4; ++j) {
    float bb = bm[(size_t)n0 + wn * 64 + j * 16 + ln];
    for (int r = 0; r < 8; ++r) {
      acc[0][j][r] += bb;
      acc[1][j][r] += bb;
    }
  }

  // RoPE (z < 2) + store.  cos/sin identical for hd and hd+32 (emb = cat(f, f)).
  const float NEG_LN_THETA_DIV = -0.28782313662425572f;  // -ln(10000)/32
  const float scale = (z == 0) ? 8.0f : 1.0f;            // sqrt(HD) folded into q
  for (int i = 0; i < 2; ++i) {
    for (int r = 0; r < 8; ++r) {
      int m = m0 + wm * 32 + i * 16 + r + 8 * lh;
      int s = m & (S - 1);
      int b = m >> 11;
      if (z < 2) {
        for (int j = 0; j < 2; ++j) {
          int hd = j * 16 + ln;  // < 32; partner hd+32 is tile j+2, same lane/reg
          float ang = (float)s * __expf((float)hd * NEG_LN_THETA_DIV);
          float sn, cs;
          __sincosf(ang, &sn, &cs);
          float lo = acc[i][j][r], hi = acc[i][j + 2][r];
          acc[i][j][r] = lo * cs - hi * sn;
          acc[i][j + 2][r] = hi * cs + lo * sn;
        }
      }
      for (int j = 0; j < 4; ++j) {
        int n = n0 + wn * 64 + j * 16 + ln;
        int hh = (n >> 6) & 15;
        int hd = n & 63;
        float val = acc[i][j][r] * scale;
        if (z == 2)
          vws[(((size_t)b * 16 + hh) * 64 + hd) * S + s] = (_Float16)val;
        else if (z == 0)
          qws[(((size_t)b * 16 + hh) * S + s) * 64 + hd] = (_Float16)val;
        else
          kws[(((size_t)b * 16 + hh) * S + s) * 64 + hd] = (_Float16)val;
      }
    }
  }
}

// ---------------------------------------------------------------------------
// Kernel 2: sliding-window flash attention.  One block = (b, h, 64 queries),
// 4 waves x 16 queries.  32-key blocks over the band, double-buffered K/V
// staged with global_load_async_to_lds_b128 (ASYNCcnt), one barrier per iter.
// ---------------------------------------------------------------------------
__global__ __launch_bounds__(128) void swa_attn_kernel(
    const _Float16* __restrict__ qg, const _Float16* __restrict__ kg,
    const _Float16* __restrict__ vg, const int* __restrict__ amask,
    float* __restrict__ out) {
  constexpr int S = 2048, WIN = 256;
  const int b = blockIdx.z, h = blockIdx.y;
  const int q0 = blockIdx.x * 64;
  const int tid = threadIdx.x;
  const int lane = tid & 31, ln = lane & 15, lh = lane >> 4;
  const int wave = tid >> 5;

  __shared__ __align__(32) _Float16 Qs[64 * 64];       // [m][hd]
  __shared__ __align__(32) _Float16 Ks[2][32 * 64];    // [key][hd]
  __shared__ __align__(32) _Float16 Vs[2][64 * 32];    // [hd][key]
  __shared__ __align__(32) _Float16 Ps[4][16 * 32];    // per-wave P [m][key]

  const size_t bhh = (size_t)b * 16 + h;
  const _Float16* qrow = qg + (bhh * S + q0) * 64;  // 64x64 contiguous slab
  const _Float16* krow = kg + bhh * S * 64;         // [s][hd]
  const _Float16* vrow = vg + bhh * 64 * S;         // [hd][s]

  const int kstart = (q0 - WIN > 0) ? (q0 - WIN) : 0;
  const int kend = (q0 + 64 + WIN < S) ? (q0 + 64 + WIN) : S;
  const int nb = (kend - kstart) >> 5;

  // Async-stage the Q slab once (waves 2,3): 8192 B = 512 x 16B chunks.
  if (wave >= 2) {
    int c0 = (wave - 2) * 256 + lane;
    for (int c = c0; c < (wave - 1) * 256; c += 32)
      async_copy16(qrow + c * 8, lds_addr(&Qs[c * 8]));
  }
  // K/V block stage: wave 0 copies K (contiguous 4096 B), wave 1 copies V
  // (64 rows x 64 B, strided in global, linear in LDS).
  auto issue_kv = [&](int kb, int buf) {
    if (wave == 0) {
      const _Float16* src = krow + (size_t)kb * 64;
      for (int c = lane; c < 256; c += 32)
        async_copy16(src + c * 8, lds_addr(&Ks[buf][c * 8]));
    } else if (wave == 1) {
      for (int c = lane; c < 256; c += 32) {
        int hd = c >> 2, part = c & 3;
        async_copy16(vrow + (size_t)hd * S + kb + part * 8,
                     lds_addr(&Vs[buf][c * 8]));
      }
    }
  };
  issue_kv(kstart, 0);

  f8 oacc[4] = {};
  float mrow[8], lrow[8];
  for (int r = 0; r < 8; ++r) { mrow[r] = -1e30f; lrow[r] = 0.0f; }

  for (int ib = 0; ib < nb; ++ib) {
    const int kb = kstart + ib * 32;
    const int cur = ib & 1;

    asm volatile("s_wait_asynccnt 0" ::: "memory");
    __syncthreads();
    if (ib + 1 < nb) issue_kv(kb + 32, cur ^ 1);  // prefetch next block

    // S = Q @ K^T (q pre-scaled by sqrt(64)); M=16/wave, N=32 keys, K=64.
    f8 sacc[2] = {};
    for (int ks2 = 0; ks2 < 2; ++ks2) {
      int mq = wave * 16 + ln;
      int kb8 = ks2 * 32 + lh * 8;
      const h8* p0 = (const h8*)&Qs[mq * 64 + kb8];
      const h8* p1 = (const h8*)&Qs[mq * 64 + kb8 + 16];
      h16 afr = join8(*p0, *p1);
      for (int t = 0; t < 2; ++t) {
        h16 bfr = join8(*(const h8*)&Ks[cur][(t * 16 + ln) * 64 + kb8], *(const h8*)&Ks[cur][(t * 16 + ln) * 64 + kb8 + 16]);
        sacc[t] = WMMA_F32_F16(afr, bfr, sacc[t]);
      }
    }

    // Band + key mask, then online softmax (rows = r + 8*lh, cols across lanes).
    const int amj0 = amask[(size_t)b * S + kb + ln];
    const int amj1 = amask[(size_t)b * S + kb + 16 + ln];
    float sv[2][8];
    for (int t = 0; t < 2; ++t)
      for (int r = 0; r < 8; ++r) {
        int iq = q0 + wave * 16 + r + 8 * lh;
        int j = kb + t * 16 + ln;
        bool ok = (j >= iq - WIN) && (j <= iq + WIN) && ((t ? amj1 : amj0) != 0);
        sv[t][r] = ok ? sacc[t][r] : -1e30f;
      }
    for (int r = 0; r < 8; ++r) {
      float mx = fmaxf(sv[0][r], sv[1][r]);
      for (int d = 1; d < 16; d <<= 1) mx = fmaxf(mx, __shfl_xor(mx, d, 32));
      float mnew = fmaxf(mrow[r], mx);
      float alpha = __expf(mrow[r] - mnew);
      mrow[r] = mnew;
      float psum = 0.0f;
      for (int t = 0; t < 2; ++t) {
        float p = (sv[t][r] <= -1e29f) ? 0.0f : __expf(sv[t][r] - mnew);
        sv[t][r] = p;
        psum += p;
      }
      for (int d = 1; d < 16; d <<= 1) psum += __shfl_xor(psum, d, 32);
      lrow[r] = lrow[r] * alpha + psum;
      for (int o = 0; o < 4; ++o) oacc[o][r] *= alpha;
    }

    // C-layout -> A-fragment layout for P via wave-private LDS (in-order DS).
    for (int t = 0; t < 2; ++t)
      for (int r = 0; r < 8; ++r)
        Ps[wave][(r + 8 * lh) * 32 + t * 16 + ln] = (_Float16)sv[t][r];
    asm volatile("s_wait_dscnt 0" ::: "memory");

    // O += P @ V  (K = 32 keys, one WMMA step; N = 64 hd = 4 tiles).
    {
      const h8* p0 = (const h8*)&Ps[wave][ln * 32 + lh * 8];
      const h8* p1 = (const h8*)&Ps[wave][ln * 32 + lh * 8 + 16];
      h16 afr = join8(*p0, *p1);
      for (int o = 0; o < 4; ++o) {
        h16 bfr = join8(*(const h8*)&Vs[cur][(o * 16 + ln) * 32 + lh * 8], *(const h8*)&Vs[cur][(o * 16 + ln) * 32 + lh * 8 + 16]);
        oacc[o] = WMMA_F32_F16(afr, bfr, oacc[o]);
      }
    }
  }

  // Normalize, apply query mask, store [B, S, H*64] fp32.
  for (int r = 0; r < 8; ++r) {
    int sq = q0 + wave * 16 + r + 8 * lh;
    int qm = amask[(size_t)b * S + sq];
    float inv = (qm != 0 && lrow[r] > 0.0f) ? 1.0f / lrow[r] : 0.0f;
    for (int o = 0; o < 4; ++o) {
      int hd = o * 16 + ln;
      out[((size_t)b * S + sq) * 1024 + h * 64 + hd] = oacc[o][r] * inv;
    }
  }
}

// ---------------------------------------------------------------------------
extern "C" void kernel_launch(void* const* d_in, const int* in_sizes, int n_in,
                              void* d_out, int out_size, void* d_ws, size_t ws_size,
                              hipStream_t stream) {
  (void)in_sizes; (void)n_in; (void)out_size; (void)ws_size;
  const float* X = (const float*)d_in[0];
  const int* amask = (const int*)d_in[1];
  const float* Wq = (const float*)d_in[2];
  const float* bq = (const float*)d_in[3];
  const float* Wk = (const float*)d_in[4];
  const float* bk = (const float*)d_in[5];
  const float* Wv = (const float*)d_in[6];
  const float* bv = (const float*)d_in[7];

  const size_t QK_ELEMS = (size_t)2 * 16 * 2048 * 64;  // B*H*S*HD
  _Float16* qws = (_Float16*)d_ws;
  _Float16* kws = qws + QK_ELEMS;
  _Float16* vws = kws + QK_ELEMS;

  dim3 g1(32, 8, 3), b1(256);
  hipLaunchKernelGGL(qkv_rope_kernel, g1, b1, 0, stream,
                     X, Wq, bq, Wk, bk, Wv, bv, qws, kws, vws);

  dim3 g2(32, 16, 2), b2(128);
  hipLaunchKernelGGL(swa_attn_kernel, g2, b2, 0, stream,
                     qws, kws, vws, amask, (float*)d_out);
}
